// GraphAttention_21792664060190
// MI455X (gfx1250) — compile-verified, matrix operand order checked by reading, not run on hardware
//
#include <hip/hip_runtime.h>

// ---------------------------------------------------------------------------
// GAT layer for MI455X (gfx1250, wave32, WMMA + TDM).
// B=256, N=199, F=256, H=8, F'=256.
//  - both big GEMMs (~95 GFLOP total) run on v_wmma_f32_16x16x32_bf16
//  - projection kernel writes feats TRANSPOSED (featsT[b,h,d,j]) so the
//    aggregation GEMM needs no LDS transpose at all
//  - aggregation stages its attn A-tile via the Tensor Data Mover
// ---------------------------------------------------------------------------

#define B_      256
#define N_      199
#define F_IN    256
#define H_      8
#define F_OUT   256
#define N_PADK  224   // K (neighbor) dim padded to 7*32 for the WMMA K-loop
#define N_PADR  208   // row dim padded to 13*16 tiles
#define NEGV    (-1e16f)

typedef __attribute__((ext_vector_type(16))) __bf16 v16bf;
typedef __attribute__((ext_vector_type(8)))  __bf16 v8bf;
typedef __attribute__((ext_vector_type(8)))  float  v8f;
typedef __attribute__((ext_vector_type(4)))  unsigned int v4u;
typedef __attribute__((ext_vector_type(8)))  int    v8i;
typedef __attribute__((ext_vector_type(4)))  int    v4i;

#if defined(__has_builtin)
#if __has_builtin(__builtin_amdgcn_tensor_load_to_lds)
#define USE_TDM 1
#endif
#endif
#if __has_include(<hip/amd_detail/amd_gfx1250_TDM.h>)
#define TDM_ARITY6 1   // therock-10.0 headers -> 6-arg builtin
#endif

// Build the 16-half WMMA operand from the two contiguous 8-half chunks that
// the CDNA5 A/B VGPR layout demands (halves 0..7 -> K+0..7, 8..15 -> K+16..23).
static __device__ inline v16bf pack16(v8bf lo, v8bf hi) {
    v16bf r;
#pragma unroll
    for (int i = 0; i < 8; ++i) { r[i] = lo[i]; r[i + 8] = hi[i]; }
    return r;
}

// ---------------------------------------------------------------------------
// K0: W[h][f][d] fp32  ->  Wt[h][d][f] bf16   (contiguous-K B-operand reads)
// ---------------------------------------------------------------------------
__global__ __launch_bounds__(256) void wt_kernel(const float* __restrict__ W,
                                                 __bf16* __restrict__ Wt) {
    int idx = blockIdx.x * 256 + threadIdx.x;       // < H*F_IN*F_OUT = 524288
    int h = idx >> 16;
    int f = (idx >> 8) & 255;
    int d = idx & 255;
    Wt[((size_t)((h << 8) | d) << 8) | f] = (__bf16)W[idx];
}

// ---------------------------------------------------------------------------
// K1: projection GEMM  featsT[b,h,d,j] = sum_f X[b,j,f] * W[h,f,d]  (bf16 out,
// stored TRANSPOSED: C-tile lane owns one d and 8 consecutive rows -> one
// 16-byte store per tile).
// ---------------------------------------------------------------------------
__global__ __launch_bounds__(256) void proj_kernel(const float* __restrict__ X,
                                                   const __bf16* __restrict__ Wt,
                                                   __bf16* __restrict__ featsT) {
    __shared__ __bf16 ldsX[16][264];                // pad 264: conflict-free rows
    const int r0 = blockIdx.x * 16;
    const int h  = blockIdx.y;
    const int b  = blockIdx.z;
    const int t  = threadIdx.x;

    // Stage X tile (16 x 256) -> bf16 LDS; rows >= N_ become zeros.
    {
        int row = t >> 4;
        int cb  = (t & 15) << 4;
        int gr  = r0 + row;
        if (gr < N_) {
            const float* src = X + ((size_t)b * N_ + gr) * F_IN + cb;
#pragma unroll
            for (int i = 0; i < 16; ++i) ldsX[row][cb + i] = (__bf16)src[i];
        } else {
#pragma unroll
            for (int i = 0; i < 16; ++i) ldsX[row][cb + i] = (__bf16)0.0f;
        }
    }
    __syncthreads();

    const int wave = t >> 5;
    const int lane = t & 31;
    const int r16  = lane & 15;
    const int hi   = lane >> 4;                     // kbase = hi*8 per ISA layout

#pragma unroll
    for (int tt = 0; tt < 2; ++tt) {                // 2 column tiles per wave
        const int n0 = (wave * 2 + tt) << 4;
        const __bf16* wcol = Wt + ((size_t)((h << 8) | (n0 + r16)) << 8);
        v8f c = {};
#pragma unroll
        for (int kk = 0; kk < 8; ++kk) {            // K = 256 in steps of 32
            const int kb = kk * 32 + hi * 8;
            v8bf alo = *(const v8bf*)&ldsX[r16][kb];
            v8bf ahi = *(const v8bf*)&ldsX[r16][kb + 16];
            v8bf blo = *(const v8bf*)&wcol[kb];
            v8bf bhi = *(const v8bf*)&wcol[kb + 16];
            c = __builtin_amdgcn_wmma_f32_16x16x32_bf16(false, pack16(alo, ahi),
                                                        false, pack16(blo, bhi),
                                                        (short)0, c, false, false);
        }
        // C layout: lane holds col d = n0+r16; VGPR r holds row r0 + hi*8 + r.
        v8bf st;
#pragma unroll
        for (int r = 0; r < 8; ++r) st[r] = (__bf16)c[r];
        *(v8bf*)(featsT + ((size_t)(b * H_ + h) * 256 + n0 + r16) * N_PADK
                        + r0 + hi * 8) = st;
    }
}

// ---------------------------------------------------------------------------
// K2: e_self / e_neigh from featsT.  Block per (b,h): thread t handles rows
// j=2t,2t+1; column-wise reads are fully coalesced u32 (two bf16) loads.
// ---------------------------------------------------------------------------
__global__ __launch_bounds__(128) void attn_e_kernel(const __bf16* __restrict__ featsT,
                                                     const float* __restrict__ aS,
                                                     const float* __restrict__ aN,
                                                     float* __restrict__ eS,
                                                     float* __restrict__ eN) {
    __shared__ float sa[256], sn[256];
    const int bh = blockIdx.x;
    const int h  = bh & (H_ - 1);
    for (int d = threadIdx.x; d < 256; d += 128) {
        sa[d] = aS[h * 256 + d];
        sn[d] = aN[h * 256 + d];
    }
    __syncthreads();
    const int t = threadIdx.x;
    if (t >= 112) return;                           // 112*2 = 224 rows
    const __bf16* base = featsT + (size_t)bh * 256 * N_PADK + 2 * t;
    float s0a = 0.f, s1a = 0.f, s0b = 0.f, s1b = 0.f;
    for (int d = 0; d < 256; ++d) {
        unsigned u = *(const unsigned*)(base + (size_t)d * N_PADK);
        float f0 = __uint_as_float((u & 0xFFFFu) << 16);   // bf16 -> f32
        float f1 = __uint_as_float(u & 0xFFFF0000u);
        float a = sa[d], n = sn[d];
        s0a += f0 * a; s0b += f1 * a;
        s1a += f0 * n; s1b += f1 * n;
    }
    int i0 = 2 * t, i1 = 2 * t + 1;
    if (i0 < N_) { eS[bh * N_PADR + i0] = s0a; eN[bh * N_PADR + i0] = s1a; }
    if (i1 < N_) { eS[bh * N_PADR + i1] = s0b; eN[bh * N_PADR + i1] = s1b; }
}

// ---------------------------------------------------------------------------
// K3: masked softmax over neighbors, one wave per (b,h,i); bf16 attn output
// padded with zeros out to K=224 (so padded featsT columns contribute nothing).
// ---------------------------------------------------------------------------
__global__ __launch_bounds__(256) void softmax_kernel(const int* __restrict__ A,
                                                      const float* __restrict__ eS,
                                                      const float* __restrict__ eN,
                                                      __bf16* __restrict__ attn) {
    int rid = blockIdx.x * 8 + (threadIdx.x >> 5);
    if (rid >= B_ * H_ * N_) return;
    int lane = threadIdx.x & 31;
    int b = rid / (H_ * N_);
    int rem = rid % (H_ * N_);
    int h = rem / N_;
    int i = rem % N_;
    int bh = b * H_ + h;
    float es = eS[bh * N_PADR + i];
    const float* en = eN + bh * N_PADR;

    float lg[7];
    float mx = -3.4e38f;
#pragma unroll
    for (int q = 0; q < 7; ++q) {                   // j = lane + 32q, covers 224
        int j = lane + q * 32;
        float v = -3.4e38f;
        if (j < N_) v = es + en[j] + (A[i * N_ + j] ? 0.0f : NEGV);
        lg[q] = v;
        mx = fmaxf(mx, v);
    }
#pragma unroll
    for (int off = 16; off >= 1; off >>= 1) mx = fmaxf(mx, __shfl_xor(mx, off, 32));
    float sum = 0.f;
#pragma unroll
    for (int q = 0; q < 7; ++q) {
        int j = lane + q * 32;
        float e = (j < N_) ? __expf(lg[q] - mx) : 0.0f;
        lg[q] = e;
        sum += e;
    }
#pragma unroll
    for (int off = 16; off >= 1; off >>= 1) sum += __shfl_xor(sum, off, 32);
    float inv = 1.0f / sum;                          // self-loop -> nonzero sum
    __bf16* arow = attn + ((size_t)bh * N_PADR + i) * N_PADK;
#pragma unroll
    for (int q = 0; q < 7; ++q) arow[lane + q * 32] = (__bf16)(lg[q] * inv);
}

// ---------------------------------------------------------------------------
// K4: aggregation GEMM  out[b,i,h*256+d] = sum_j attn[b,h,i,j]*featsT[b,h,d,j]
// attn tile (16x224) staged once into LDS by the Tensor Data Mover; B-operand
// comes straight from featsT (K-contiguous 16B loads, L2-resident per (b,h)).
// Fully unrolled K-loop: no barriers inside, pipelines like K1.
// ---------------------------------------------------------------------------
__global__ __launch_bounds__(256) void agg_kernel(const __bf16* __restrict__ attn,
                                                  const __bf16* __restrict__ featsT,
                                                  const float* __restrict__ bias,
                                                  float* __restrict__ out) {
    __shared__ __bf16 ldsA[16 * N_PADK];            // attn tile, row-major [i][j]
    const int i0 = blockIdx.x * 16;
    const int h  = blockIdx.y;
    const int b  = blockIdx.z;
    const int bh = b * H_ + h;

#if defined(USE_TDM)
    if (threadIdx.x < 32) {                         // wave 0 issues the TDM op
        unsigned long long gaddr =
            (unsigned long long)(const void*)(attn + ((size_t)bh * N_PADR + i0) * N_PADK);
        unsigned ldsOff = (unsigned)(unsigned long long)(const void*)&ldsA[0];
        // D# group0: count=1 | lds_addr | global_addr(57b) | type=2
        v4u g0 = { 1u,
                   ldsOff,
                   (unsigned)(gaddr & 0xFFFFFFFFu),
                   (unsigned)((gaddr >> 32) & 0x01FFFFFFu) | (2u << 30) };
        // D# group1: data_size=2B; tensor 224 x 208 (stride 224); tile 224 x 16
        v8i g1 = { (int)(1u << 16),                 // data_size code 1 (2 bytes)
                   (int)((unsigned)N_PADK << 16),   // tensor_dim0 = 224 (bits 79:48)
                   (int)((unsigned)N_PADR << 16),   // tensor_dim1 = 208 (bits 111:80)
                   (int)((unsigned)N_PADK << 16),   // tile_dim0 = 224 (bits 127:112)
                   16,                              // tile_dim1 = 16, tile_dim2 = 0
                   N_PADK,                          // tensor_dim0_stride = 224
                   0, 0 };
        v4i g2 = { 0, 0, 0, 0 };                    // dims 2/3 unused (tile_dim2=0)
        v4i g3 = { 0, 0, 0, 0 };
#if defined(TDM_ARITY6)
        v8i g4 = { 0, 0, 0, 0, 0, 0, 0, 0 };
        __builtin_amdgcn_tensor_load_to_lds(g0, g1, g2, g3, g4, 0);
#else
        __builtin_amdgcn_tensor_load_to_lds(g0, g1, g2, g3, 0);
#endif
        __builtin_amdgcn_s_wait_tensorcnt(0);
    }
    __syncthreads();
#else
    // Fallback: cooperative copy of the attn tile.
#pragma unroll
    for (int q = 0; q < 2; ++q) {
        int cidx = threadIdx.x + q * 256;           // 448 chunks of 8 bf16
        if (cidx < 448) {
            int row = cidx / 28;
            int cb  = (cidx % 28) * 8;
            *(v8bf*)&ldsA[row * N_PADK + cb] =
                *(const v8bf*)(attn + ((size_t)bh * N_PADR + i0 + row) * N_PADK + cb);
        }
    }
    __syncthreads();
#endif

    const int t    = threadIdx.x;
    const int wave = t >> 5;
    const int lane = t & 31;
    const int r16  = lane & 15;
    const int hi   = lane >> 4;
    const int d0a  = wave * 32;
    const int d0b  = wave * 32 + 16;

    const __bf16* arow = ldsA + r16 * N_PADK;
    const __bf16* fa = featsT + ((size_t)bh * 256 + d0a + r16) * N_PADK;
    const __bf16* fb = featsT + ((size_t)bh * 256 + d0b + r16) * N_PADK;
    v8f c0 = {}, c1 = {};

#pragma unroll
    for (int kk = 0; kk < 7; ++kk) {                // K = 224 in steps of 32
        const int kb = kk * 32 + hi * 8;
        v8bf alo = *(const v8bf*)&arow[kb];
        v8bf ahi = *(const v8bf*)&arow[kb + 16];
        v16bf a = pack16(alo, ahi);
        {
            v8bf blo = *(const v8bf*)&fa[kb];
            v8bf bhi = *(const v8bf*)&fa[kb + 16];
            c0 = __builtin_amdgcn_wmma_f32_16x16x32_bf16(false, a, false,
                                                         pack16(blo, bhi),
                                                         (short)0, c0, false, false);
        }
        {
            v8bf blo = *(const v8bf*)&fb[kb];
            v8bf bhi = *(const v8bf*)&fb[kb + 16];
            c1 = __builtin_amdgcn_wmma_f32_16x16x32_bf16(false, a, false,
                                                         pack16(blo, bhi),
                                                         (short)0, c1, false, false);
        }
    }

    float bv0 = bias[h * F_OUT + d0a + r16];
    float bv1 = bias[h * F_OUT + d0b + r16];
#pragma unroll
    for (int r = 0; r < 8; ++r) {
        int i = i0 + hi * 8 + r;
        if (i < N_) {
            float* orow = out + ((size_t)b * N_ + i) * (H_ * F_OUT) + h * F_OUT;
            orow[d0a + r16] = c0[r] + bv0;
            orow[d0b + r16] = c1[r] + bv1;
        }
    }
}

// ---------------------------------------------------------------------------
// Launch: X, A, W, a_self, a_neigh, bias  (setup_inputs order)
// ---------------------------------------------------------------------------
extern "C" void kernel_launch(void* const* d_in, const int* in_sizes, int n_in,
                              void* d_out, int out_size, void* d_ws, size_t ws_size,
                              hipStream_t stream) {
    (void)in_sizes; (void)n_in; (void)out_size; (void)ws_size;
    const float* X    = (const float*)d_in[0];
    const int*   A    = (const int*)d_in[1];
    const float* W    = (const float*)d_in[2];
    const float* aS   = (const float*)d_in[3];
    const float* aN   = (const float*)d_in[4];
    const float* bias = (const float*)d_in[5];
    float* out = (float*)d_out;

    char* ws = (char*)d_ws;
    const size_t featsBytes = (size_t)B_ * H_ * 256 * N_PADK * 2;    // ~235 MB
    const size_t attnBytes  = (size_t)B_ * H_ * N_PADR * N_PADK * 2; // ~191 MB
    const size_t wtBytes    = (size_t)H_ * F_IN * F_OUT * 2;         //   1 MB
    const size_t eBytes     = (size_t)B_ * H_ * N_PADR * 4;          // 1.7 MB

    __bf16* featsT = (__bf16*)ws;
    __bf16* attn   = (__bf16*)(ws + featsBytes);
    __bf16* Wt     = (__bf16*)(ws + featsBytes + attnBytes);
    float*  eS     = (float*)(ws + featsBytes + attnBytes + wtBytes);
    float*  eN     = (float*)(ws + featsBytes + attnBytes + wtBytes + eBytes);

    wt_kernel<<<(H_ * F_IN * F_OUT) / 256, 256, 0, stream>>>(W, Wt);
    proj_kernel<<<dim3(13, H_, B_), 256, 0, stream>>>(X, Wt, featsT);

    attn_e_kernel<<<B_ * H_, 128, 0, stream>>>(featsT, aS, aN, eS, eN);

    const int rows = B_ * H_ * N_;                  // 407552, divisible by 8
    softmax_kernel<<<(rows + 7) / 8, 256, 0, stream>>>(A, eS, eN, attn);

    agg_kernel<<<dim3(13, H_, B_), 256, 0, stream>>>(attn, featsT, bias, out);
}
